// CrossLayerTranscoderSystem_46686294507773
// MI455X (gfx1250) — matrix operand order repaired and checked by reading, not compiled, not run on hardware
//
#include <hip/hip_runtime.h>

// Problem constants (from reference setup_inputs)
#define L_ 6
#define N_ 1024          // B*S = 4*256
#define D_ 768
#define H_ 4096
#define O_ 768
#define K_ 64
#define NPAIR 21         // number of (s,t) pairs with s <= t

typedef __attribute__((ext_vector_type(16))) __bf16 v16bf;
typedef __attribute__((ext_vector_type(8)))  __bf16 v8bf;
typedef __attribute__((ext_vector_type(8)))  float  v8f;

// ---------------------------------------------------------------------------
// Kernel 0: split fp32 -> (bf16 hi, bf16 lo) planes, done ONCE per element so
// the GEMM hot loop carries zero conversion VALU work.
// ---------------------------------------------------------------------------
__global__ __launch_bounds__(256) void split_bf16_kernel(
    const float* __restrict__ src, __bf16* __restrict__ hi,
    __bf16* __restrict__ lo, long n)
{
    long i = (long)blockIdx.x * 256 + threadIdx.x;
    if (i < n) {
        float x = src[i];
        __bf16 h = (__bf16)x;
        hi[i] = h;
        lo[i] = (__bf16)(x - (float)h);
    }
}

// ---------------------------------------------------------------------------
// Kernel 1: encode GEMM  pre[l] = X[l] @ Wenc[l]^T + bias, fused JumpReLU.
// Each wave computes a 32(M) x 32(N) block as 2x2 16x16 WMMA tiles.
// fp32 accuracy via 3-term bf16 split: hi*hi + hi*lo + lo*hi (fp32 accum).
// Inner loop: 16x b128 loads -> 12x v_wmma_f32_16x16x32_bf16.
// ---------------------------------------------------------------------------
__global__ __launch_bounds__(256) void encode_kernel(
    const __bf16* __restrict__ Xh, const __bf16* __restrict__ Xl,  // [L,N,D]
    const __bf16* __restrict__ Wh, const __bf16* __restrict__ Wl,  // [L,H,D]
    const float* __restrict__ bEnc,  // [L, H]
    const float* __restrict__ thr,   // [L, H]
    float* __restrict__ preF)        // [L, N, H]  (JumpReLU-masked pre-acts)
{
    const int lane   = threadIdx.x & 31;
    const int waveId = threadIdx.x >> 5;
    long unit = (long)blockIdx.x * 8 + waveId;   // grid sized exactly

    const int unitsH = H_ / 32;                  // 128
    const int unitsN = N_ / 32;                  // 32
    const int uh = (int)(unit % unitsH); unit /= unitsH;
    const int un = (int)(unit % unitsN); unit /= unitsN;
    const int l  = (int)unit;                    // 0..5

    const int m0 = un * 32;
    const int h0 = uh * 32;
    const int lane15 = lane & 15;
    const int hiHalf = lane >> 4;                // 0 or 1

    const __bf16* aH[2]; const __bf16* aL[2];
    const __bf16* bH[2]; const __bf16* bL[2];
    #pragma unroll
    for (int i = 0; i < 2; ++i) {
        const long ao = ((long)l * N_ + (m0 + 16 * i + lane15)) * D_;
        aH[i] = Xh + ao;  aL[i] = Xl + ao;
        const long bo = ((long)l * H_ + (h0 + 16 * i + lane15)) * D_;
        bH[i] = Wh + bo;  bL[i] = Wl + bo;
    }

    v8f acc[4] = {{}, {}, {}, {}};
    for (int d0 = 0; d0 < D_; d0 += 32) {
        // A fragment (this code's contiguous k order, same as B; not the ISA table order):
        //   half 0: K = 0..15 ; half 1: K = 16..31
        v16bf ah[2], al[2], bh[2], bl[2];
        #pragma unroll
        for (int i = 0; i < 2; ++i) {
            v8bf h0v = *(const v8bf*)(aH[i] + d0 + hiHalf * 16);
            v8bf h1v = *(const v8bf*)(aH[i] + d0 + hiHalf * 16 + 8);
            ah[i] = __builtin_shufflevector(h0v, h1v,
                     0, 1, 2, 3, 4, 5, 6, 7, 8, 9, 10, 11, 12, 13, 14, 15);
            v8bf l0v = *(const v8bf*)(aL[i] + d0 + hiHalf * 16);
            v8bf l1v = *(const v8bf*)(aL[i] + d0 + hiHalf * 16 + 8);
            al[i] = __builtin_shufflevector(l0v, l1v,
                     0, 1, 2, 3, 4, 5, 6, 7, 8, 9, 10, 11, 12, 13, 14, 15);
            // B fragment (32x16): lanes 0-15 hold K=0..15, lanes 16-31 K=16..31
            bh[i] = *(const v16bf*)(bH[i] + d0 + hiHalf * 16);
            bl[i] = *(const v16bf*)(bL[i] + d0 + hiHalf * 16);
        }
        #pragma unroll
        for (int mi = 0; mi < 2; ++mi) {
            #pragma unroll
            for (int hi = 0; hi < 2; ++hi) {
                v8f c = acc[mi * 2 + hi];
                c = __builtin_amdgcn_wmma_f32_16x16x32_bf16(false, ah[mi], false, bh[hi],
                                                            (short)0, c, false, false);
                c = __builtin_amdgcn_wmma_f32_16x16x32_bf16(false, ah[mi], false, bl[hi],
                                                            (short)0, c, false, false);
                c = __builtin_amdgcn_wmma_f32_16x16x32_bf16(false, al[mi], false, bh[hi],
                                                            (short)0, c, false, false);
                acc[mi * 2 + hi] = c;
            }
        }
    }

    // C/D layout: lane (0-15) = column N, VGPR j = row M=j (half 0) / M=j+8 (half 1)
    #pragma unroll
    for (int hi = 0; hi < 2; ++hi) {
        const int h  = h0 + hi * 16 + lane15;
        const float bv = bEnc[l * H_ + h];
        const float tv = thr [l * H_ + h];
        #pragma unroll
        for (int mi = 0; mi < 2; ++mi) {
            v8f c = acc[mi * 2 + hi];
            #pragma unroll
            for (int j = 0; j < 8; ++j) {
                const int m = m0 + mi * 16 + j + 8 * hiHalf;
                float p = c[j] + bv;
                preF[((long)l * N_ + m) * H_ + h] = (p > tv) ? p : 0.0f;
            }
        }
    }
}

// ---------------------------------------------------------------------------
// Kernel 2: per-row top-K via 4-pass 8-bit radix select in LDS.
// One 256-thread block per row (L*N = 6144 rows). Order within the top-K is
// irrelevant (reference scatters back into a dense vector).
// ---------------------------------------------------------------------------
__global__ __launch_bounds__(256) void topk_kernel(
    const float* __restrict__ preF,  // [L*N, H]
    float* __restrict__ tkV,         // [L*N, K]
    int*   __restrict__ tkI)         // [L*N, K]
{
    __shared__ unsigned keys[H_];    // 16 KB
    __shared__ unsigned hist[256];
    __shared__ unsigned sPrefix, sKrem, sCnt1, sCnt2;

    const int row = blockIdx.x;
    const int tid = threadIdx.x;
    const float* src = preF + (long)row * H_;

    // order-preserving float -> u32 map
    for (int i = tid; i < H_; i += 256) {
        unsigned u = __float_as_uint(src[i]);
        keys[i] = (u & 0x80000000u) ? ~u : (u | 0x80000000u);
    }
    if (tid == 0) { sCnt1 = 0; sCnt2 = 0; }
    __syncthreads();

    unsigned prefix = 0;
    unsigned kRem = K_;
    for (int pass = 0; pass < 4; ++pass) {
        const int shift = 24 - pass * 8;
        const unsigned mask = (pass == 0) ? 0u : (0xFFFFFFFFu << (shift + 8));

        hist[tid] = 0;
        __syncthreads();
        for (int i = tid; i < H_; i += 256) {
            unsigned k = keys[i];
            if ((k & mask) == prefix)
                atomicAdd(&hist[(k >> shift) & 0xFFu], 1u);
        }
        __syncthreads();
        if (tid == 0) {
            unsigned accum = 0;
            int b = 255;
            for (; b > 0; --b) {
                if (accum + hist[b] >= kRem) break;
                accum += hist[b];
            }
            sPrefix = prefix | ((unsigned)b << shift);
            sKrem   = kRem - accum;
        }
        __syncthreads();
        prefix = sPrefix;
        kRem   = sKrem;
        __syncthreads();
    }
    // prefix == key value of the K-th largest element

    float* vOut = tkV + (long)row * K_;
    int*   iOut = tkI + (long)row * K_;

    for (int i = tid; i < H_; i += 256) {
        if (keys[i] > prefix) {
            unsigned slot = atomicAdd(&sCnt1, 1u);
            vOut[slot] = src[i];
            iOut[slot] = i;
        }
    }
    __syncthreads();
    const unsigned nGreater = sCnt1;     // < K by construction
    for (int i = tid; i < H_; i += 256) {
        if (keys[i] == prefix) {
            unsigned slot = atomicAdd(&sCnt2, 1u);
            if (slot < (unsigned)K_ - nGreater) {
                vOut[nGreater + slot] = src[i];
                iOut[nGreater + slot] = i;
            }
        }
    }
}

// ---------------------------------------------------------------------------
// Kernel 3: transpose the 21 used decoder slabs [O,H] -> [H,O] so that sparse
// decode gathers become contiguous 3 KB row reads.
// ---------------------------------------------------------------------------
__global__ __launch_bounds__(256) void transpose_dec_kernel(
    const float* __restrict__ Wdec,  // [L, L, O, H]
    float* __restrict__ decT)        // [NPAIR, H, O]
{
    __shared__ float tile[32][33];
    const int pair = blockIdx.z;
    int t = 0;
    while ((t + 1) * (t + 2) / 2 <= pair) ++t;
    const int s = pair - t * (t + 1) / 2;

    const float* srcBase = Wdec + (((long)s * L_ + t) * O_) * H_;
    const int h0 = blockIdx.x * 32;
    const int o0 = blockIdx.y * 32;
    const int lx = threadIdx.x & 31;
    const int ly = threadIdx.x >> 5;     // 0..7

    #pragma unroll
    for (int r = 0; r < 32; r += 8)
        tile[ly + r][lx] = srcBase[(long)(o0 + ly + r) * H_ + (h0 + lx)];
    __syncthreads();

    float* dst = decT + ((long)pair * H_ + h0) * O_ + o0;
    #pragma unroll
    for (int r = 0; r < 32; r += 8)
        dst[(long)(ly + r) * O_ + lx] = tile[lx][ly + r];
}

// ---------------------------------------------------------------------------
// Kernel 4: sparse decode.  out[t,n,:] = out_bias[t] +
//           sum_{s<=t} sum_j tkV[s,n,j] * decT[pair(s,t), tkI[s,n,j], :]
// One block per (t, n); 3 output floats per thread in registers.
// ---------------------------------------------------------------------------
__global__ __launch_bounds__(256) void decode_kernel(
    const float* __restrict__ tkV,   // [L, N, K]
    const int*   __restrict__ tkI,   // [L, N, K]
    const float* __restrict__ decT,  // [NPAIR, H, O]
    const float* __restrict__ bOut,  // [L, O]
    float* __restrict__ out)         // [L, N, O]
{
    __shared__ float sV[K_];
    __shared__ int   sI[K_];
    const int n   = blockIdx.x;
    const int t   = blockIdx.y;
    const int tid = threadIdx.x;

    float acc0 = bOut[t * O_ + tid];
    float acc1 = bOut[t * O_ + tid + 256];
    float acc2 = bOut[t * O_ + tid + 512];

    for (int s = 0; s <= t; ++s) {
        __syncthreads();
        if (tid < K_) {
            sV[tid] = tkV[((long)s * N_ + n) * K_ + tid];
            sI[tid] = tkI[((long)s * N_ + n) * K_ + tid];
        }
        __syncthreads();
        const int pair = t * (t + 1) / 2 + s;
        const float* W = decT + (long)pair * H_ * O_;
        #pragma unroll 4
        for (int j = 0; j < K_; ++j) {
            const float v = sV[j];
            const float* wr = W + (long)sI[j] * O_;
            acc0 = fmaf(v, wr[tid],       acc0);
            acc1 = fmaf(v, wr[tid + 256], acc1);
            acc2 = fmaf(v, wr[tid + 512], acc2);
        }
    }
    float* dst = out + ((long)t * N_ + n) * O_;
    dst[tid]       = acc0;
    dst[tid + 256] = acc1;
    dst[tid + 512] = acc2;
}

// ---------------------------------------------------------------------------
// Launch. Workspace layout (256B-aligned chunks), total ~463 MB:
//   Xh/Xl [L,N,D] bf16 (9.4 MB each) | Wh/Wl [L,H,D] bf16 (37.7 MB each)
//   preF [L,N,H] f32 100.7 MB | tkV/tkI [L,N,K] 1.5 MB each
//   decT [NPAIR,H,O] f32 264.2 MB
// ---------------------------------------------------------------------------
extern "C" void kernel_launch(void* const* d_in, const int* in_sizes, int n_in,
                              void* d_out, int out_size, void* d_ws, size_t ws_size,
                              hipStream_t stream) {
    const float* X    = (const float*)d_in[0];   // layer_inputs [L,B,S,D]
    const float* Wenc = (const float*)d_in[1];   // enc_weight   [L,H,D]
    const float* bEnc = (const float*)d_in[2];   // enc_bias     [L,H]
    const float* thr  = (const float*)d_in[3];   // threshold    [L,H]
    const float* Wdec = (const float*)d_in[4];   // dec_weight   [L,L,O,H]
    const float* bOut = (const float*)d_in[5];   // out_bias     [L,O]
    float* out = (float*)d_out;                  // [L,B,S,O]

    const long nX = (long)L_ * N_ * D_;          // 4,718,592
    const long nW = (long)L_ * H_ * D_;          // 18,874,368

    char* ws = (char*)d_ws;
    auto take = [&](size_t bytes) {
        char* p = ws;
        ws += (bytes + 255) & ~(size_t)255;
        return p;
    };
    __bf16* Xh  = (__bf16*)take(sizeof(__bf16) * (size_t)nX);
    __bf16* Xl  = (__bf16*)take(sizeof(__bf16) * (size_t)nX);
    __bf16* Wh  = (__bf16*)take(sizeof(__bf16) * (size_t)nW);
    __bf16* Wl  = (__bf16*)take(sizeof(__bf16) * (size_t)nW);
    float*  preF = (float*)take(sizeof(float) * (size_t)L_ * N_ * H_);
    float*  tkV  = (float*)take(sizeof(float) * (size_t)L_ * N_ * K_);
    int*    tkI  = (int*)  take(sizeof(int)   * (size_t)L_ * N_ * K_);
    float*  decT = (float*)take(sizeof(float) * (size_t)NPAIR * H_ * O_);

    // 0) one-time fp32 -> bf16 hi/lo split for A and B operands
    split_bf16_kernel<<<(int)((nX + 255) / 256), 256, 0, stream>>>(X, Xh, Xl, nX);
    split_bf16_kernel<<<(int)((nW + 255) / 256), 256, 0, stream>>>(Wenc, Wh, Wl, nW);

    // 1) encode: L*(N/32)*(H/32) = 24576 wave-units, 8 waves per block
    encode_kernel<<<24576 / 8, 256, 0, stream>>>(Xh, Xl, Wh, Wl, bEnc, thr, preF);

    // 2) top-K select per row
    topk_kernel<<<L_ * N_, 256, 0, stream>>>(preF, tkV, tkI);

    // 3) transpose used decoder slabs
    dim3 tg(H_ / 32, O_ / 32, NPAIR);
    transpose_dec_kernel<<<tg, 256, 0, stream>>>(Wdec, decT);

    // 4) sparse decode
    dim3 dg(N_, L_, 1);
    decode_kernel<<<dg, 256, 0, stream>>>(tkV, tkI, decT, bOut, out);
}
